// GATModel_20126216749362
// MI455X (gfx1250) — matrix operand order repaired and checked by reading, not compiled, not run on hardware
//
#include <hip/hip_runtime.h>
#include <hip/hip_bf16.h>
#include <cmath>

typedef __attribute__((ext_vector_type(16))) _Float16 v16h;
typedef __attribute__((ext_vector_type(8)))  float    v8f;

#define NNODES 50000
#define NEDGES 800000
#define NGRAPH 128
#define NEG_SLOPE 0.2f

// ---------------------------------------------------------------------------
// WMMA GEMM: H[N x F] = X[N x K] @ W[K x F], fp32 in/out, f16 multiply, f32 acc.
// Block = F/16 waves; wave w computes the 16x16 tile at (blockIdx.x, w).
// The 16xK A panel is staged once per block into LDS, converted to f16 and
// pre-swizzled into WMMA-fragment order so each lane's per-step fragment is a
// single contiguous 32B ds_load. K-loop fully unrolled -> K/32 static wmmas.
// ---------------------------------------------------------------------------
template <int K, int F>
__global__ void gemm_wmma_f16(const float* __restrict__ X,
                              const float* __restrict__ W,
                              float* __restrict__ H) {
  constexpr int KT = K / 32;               // number of k-steps
  __shared__ v16h As[KT * 32];             // A fragments: [t][lane], 32B each

  const int wave = threadIdx.x >> 5;
  const int lane = threadIdx.x & 31;
  const int rowBase = blockIdx.x * 16;
  const int colBase = wave * 16;
  const int lmod = lane & 15;
  const int lhi  = lane >> 4;              // 0 or 1

  // Cooperative, K-coalesced load of the A panel with fragment swizzle.
  // ISA A-layout inverse: kr=k%32 -> laneHi=(kr>>3)&1, elem=(kr&7)+8*(kr>>4).
  _Float16* Ah = (_Float16*)As;
  for (int idx = threadIdx.x; idx < 16 * K; idx += blockDim.x) {
    const int row = idx / K;
    const int k   = idx - row * K;
    const float v = X[(size_t)(rowBase + row) * K + k];
    const int t   = k >> 5;
    const int kr  = k & 31;
    const int fl  = (kr >> 3) & 1;
    const int e   = (kr & 7) + ((kr >> 4) << 3);
    Ah[((t * 32) + row + 16 * fl) * 16 + e] = (_Float16)v;
  }
  __syncthreads();

  v8f acc = {};
#pragma unroll
  for (int t = 0; t < KT; ++t) {
    const v16h a = As[t * 32 + lane];      // one wide LDS read per step
    v16h b;
    const int k0 = t * 32;
#pragma unroll
    for (int e = 0; e < 16; ++e) {
      // B: 32x16 f16, lane = col (L%16); element e -> K = (e&7) + 8*(L>=16) + 16*(e>>3)
      const int kb = k0 + (e & 7) + 8 * lhi + 16 * (e >> 3);
      b[e] = (_Float16)W[(size_t)kb * F + colBase + lmod];
    }
    acc = __builtin_amdgcn_wmma_f32_16x16x32_f16(
        /*neg_a=*/false, a, /*neg_b=*/false, b,
        /*c_mod=*/(short)0, acc, /*reuse_a=*/false, /*reuse_b=*/false);
  }

#pragma unroll
  for (int r = 0; r < 8; ++r) {
    const int row = rowBase + r + 8 * lhi;  // C/D layout: VGPR r -> M = r + 8*(L>=16)
    H[(size_t)row * F + colBase + lmod] = acc[r];
  }
}

// ---------------------------------------------------------------------------
// Per-node attention scores: ssrc[i] = h[i].a_src, sdst[i] = h[i].a_dst
// one wave per node, lane-strided dot + shuffle reduce (wave32).
// ---------------------------------------------------------------------------
__global__ void scores_kernel(const float* __restrict__ H,
                              const float* __restrict__ asrc,
                              const float* __restrict__ adst,
                              float* __restrict__ ssrc, float* __restrict__ sdst,
                              int N, int F) {
  const int wave = threadIdx.x >> 5;
  const int lane = threadIdx.x & 31;
  const int node = blockIdx.x * (blockDim.x >> 5) + wave;
  if (node >= N) return;
  const float* h = H + (size_t)node * F;
  float ss = 0.f, sd = 0.f;
  for (int f = lane; f < F; f += 32) {
    const float v = h[f];
    ss += v * asrc[f];
    sd += v * adst[f];
  }
  for (int off = 16; off; off >>= 1) {
    ss += __shfl_down(ss, off, 32);
    sd += __shfl_down(sd, off, 32);
  }
  if (lane == 0) { ssrc[node] = ss; sdst[node] = sd; }
}

// ---------------------------------------------------------------------------
// Edge helpers. Edge i >= E is the implicit self-loop for node i-E.
// ---------------------------------------------------------------------------
__device__ __forceinline__ void edge_ids(int i, const int* __restrict__ src,
                                         const int* __restrict__ dst, int E,
                                         int& s, int& d) {
  if (i < E) { s = src[i]; d = dst[i]; }
  else       { s = i - E;  d = i - E;  }
}
__device__ __forceinline__ float lrelu(float v) {
  return v > 0.f ? v : NEG_SLOPE * v;
}

__global__ void edge_max_kernel(const int* __restrict__ src, const int* __restrict__ dst,
                                const float* __restrict__ ssrc, const float* __restrict__ sdst,
                                float* __restrict__ m, int E, int Ep) {
  const int i = blockIdx.x * blockDim.x + threadIdx.x;
  if (i >= Ep) return;
  int s, d; edge_ids(i, src, dst, E, s, d);
  atomicMax(m + d, lrelu(ssrc[s] + sdst[d]));      // global_atomic_max_num_f32
}

__global__ void edge_exp_kernel(const int* __restrict__ src, const int* __restrict__ dst,
                                const float* __restrict__ ssrc, const float* __restrict__ sdst,
                                const float* __restrict__ m,
                                float* __restrict__ ex, float* __restrict__ denom,
                                int E, int Ep) {
  const int i = blockIdx.x * blockDim.x + threadIdx.x;
  if (i >= Ep) return;
  int s, d; edge_ids(i, src, dst, E, s, d);
  const float v = __expf(lrelu(ssrc[s] + sdst[d]) - m[d]);
  ex[i] = v;
  atomicAdd(denom + d, v);
}

// alpha-weighted scatter:  agg[dst] += h[src] * alpha.  F/4 threads per edge,
// each does one float4 gather + 4 fp32 atomic adds.
__global__ void edge_agg_kernel(const int* __restrict__ src, const int* __restrict__ dst,
                                const float* __restrict__ ex, const float* __restrict__ denom,
                                const float* __restrict__ H, float* __restrict__ agg,
                                int E, int Ep, int F) {
  const int tpe = F >> 2;
  const long long gid = (long long)blockIdx.x * blockDim.x + threadIdx.x;
  const int i = (int)(gid / tpe);
  if (i >= Ep) return;
  const int c = (int)(gid - (long long)i * tpe) << 2;
  int s, d; edge_ids(i, src, dst, E, s, d);
  const float alpha = ex[i] / denom[d];
  const float4 hv = *(const float4*)(H + (size_t)s * F + c);
  float* out = agg + (size_t)d * F + c;
  atomicAdd(out + 0, hv.x * alpha);
  atomicAdd(out + 1, hv.y * alpha);
  atomicAdd(out + 2, hv.z * alpha);
  atomicAdd(out + 3, hv.w * alpha);
}

// ---------------------------------------------------------------------------
// small utility kernels
// ---------------------------------------------------------------------------
__global__ void fill_kernel(float* __restrict__ p, float v, size_t n) {
  size_t i = (size_t)blockIdx.x * blockDim.x + threadIdx.x;
  const size_t stride = (size_t)gridDim.x * blockDim.x;
  for (; i < n; i += stride) p[i] = v;
}

__global__ void bias_init_kernel(float* __restrict__ p, const float* __restrict__ bias,
                                 size_t n, int F) {
  size_t i = (size_t)blockIdx.x * blockDim.x + threadIdx.x;
  const size_t stride = (size_t)gridDim.x * blockDim.x;
  for (; i < n; i += stride) p[i] = bias[i % F];
}

__global__ void relu_kernel(float* __restrict__ p, size_t n) {
  size_t i = (size_t)blockIdx.x * blockDim.x + threadIdx.x;
  const size_t stride = (size_t)gridDim.x * blockDim.x;
  for (; i < n; i += stride) p[i] = fmaxf(p[i], 0.f);
}

__global__ void pool_sum_kernel(const float* __restrict__ x, const int* __restrict__ batch,
                                float* __restrict__ sums, float* __restrict__ cnt,
                                int N, int F) {
  const long long gid = (long long)blockIdx.x * blockDim.x + threadIdx.x;
  const int node = (int)(gid / F);
  if (node >= N) return;
  const int f = (int)(gid - (long long)node * F);
  const int g = batch[node];
  atomicAdd(&sums[(size_t)g * F + f], x[(size_t)node * F + f]);
  if (f == 0) atomicAdd(&cnt[g], 1.0f);
}

__global__ void pool_div_kernel(const float* __restrict__ sums, const float* __restrict__ cnt,
                                float* __restrict__ pooled, int G, int F) {
  const int gid = blockIdx.x * blockDim.x + threadIdx.x;
  if (gid >= G * F) return;
  pooled[gid] = sums[gid] / fmaxf(cnt[gid / F], 1.0f);
}

__global__ void classifier_kernel(const float* __restrict__ xs, const float* __restrict__ xt,
                                  const float* __restrict__ Wl, const float* __restrict__ bl,
                                  float* __restrict__ out, int G, int Fin, int L) {
  const int gid = blockIdx.x * blockDim.x + threadIdx.x;
  if (gid >= G * L) return;
  const int g = gid / L, l = gid - (gid / L) * L;
  float acc = bl[l];
  for (int f = 0; f < Fin; ++f)
    acc += (xs[g * Fin + f] + xt[g * Fin + f]) * Wl[(size_t)f * L + l];
  out[gid] = 1.0f / (1.0f + __expf(-acc));
}

// ---------------------------------------------------------------------------
// host orchestration
// ---------------------------------------------------------------------------
static inline unsigned cdiv(long long a, long long b) { return (unsigned)((a + b - 1) / b); }
static inline unsigned grd(long long n) {
  unsigned b = cdiv(n, 256);
  return b > 65535u ? 65535u : b;
}

extern "C" void kernel_launch(void* const* d_in, const int* in_sizes, int n_in,
                              void* d_out, int out_size, void* d_ws, size_t ws_size,
                              hipStream_t stream) {
  const int N = NNODES, E = NEDGES, G = NGRAPH, L = 1024;
  const int Ep = E + N;

  const float* x_s  = (const float*)d_in[0];
  const float* x_t  = (const float*)d_in[1];
  const int*   ei_s = (const int*)d_in[2];
  const int*   ei_t = (const int*)d_in[3];
  const int*   bs   = (const int*)d_in[4];
  const int*   bt   = (const int*)d_in[5];
  const float* Ws1 = (const float*)d_in[6],  *as1s = (const float*)d_in[7],
             * as1d = (const float*)d_in[8], *bs1 = (const float*)d_in[9];
  const float* Ws2 = (const float*)d_in[10], *as2s = (const float*)d_in[11],
             * as2d = (const float*)d_in[12],*bs2 = (const float*)d_in[13];
  const float* Wt1 = (const float*)d_in[14], *at1s = (const float*)d_in[15],
             * at1d = (const float*)d_in[16],*bt1 = (const float*)d_in[17];
  const float* Wt2 = (const float*)d_in[18], *at2s = (const float*)d_in[19],
             * at2d = (const float*)d_in[20],*bt2 = (const float*)d_in[21];
  const float* Wlin = (const float*)d_in[22], *blin = (const float*)d_in[23];

  float* ws = (float*)d_ws;
  size_t off = 0;
  auto alloc = [&](size_t n) { float* p = ws + off; off += n; return p; };
  float* A     = alloc((size_t)N * 128);  // GEMM output h (reused both layers)
  float* B     = alloc((size_t)N * 128);  // layer-1 aggregate / layer-2 input
  float* C     = alloc((size_t)N * 64);   // layer-2 aggregate
  float* ssrc  = alloc(N);
  float* sdst  = alloc(N);
  float* mx    = alloc(N);
  float* denom = alloc(N);
  float* ex    = alloc(Ep);
  float* sums  = alloc((size_t)G * 64);
  float* cnt   = alloc(G);
  float* pooled_s = alloc((size_t)G * 64);
  float* pooled_t = alloc((size_t)G * 64);

  auto gat_layer = [&](const float* xin, const int* src, const int* dst,
                       const float* W, const float* a_s, const float* a_d,
                       const float* bias, float* h, float* agg, int K, int F) {
    if (K == 256)
      gemm_wmma_f16<256, 128><<<N / 16, 256, 0, stream>>>(xin, W, h);
    else
      gemm_wmma_f16<128, 64><<<N / 16, 128, 0, stream>>>(xin, W, h);
    scores_kernel<<<cdiv(N, 8), 256, 0, stream>>>(h, a_s, a_d, ssrc, sdst, N, F);
    fill_kernel<<<grd(N), 256, 0, stream>>>(mx, -INFINITY, N);
    fill_kernel<<<grd(N), 256, 0, stream>>>(denom, 0.f, N);
    bias_init_kernel<<<grd((long long)N * F), 256, 0, stream>>>(agg, bias, (size_t)N * F, F);
    edge_max_kernel<<<cdiv(Ep, 256), 256, 0, stream>>>(src, dst, ssrc, sdst, mx, E, Ep);
    edge_exp_kernel<<<cdiv(Ep, 256), 256, 0, stream>>>(src, dst, ssrc, sdst, mx, ex, denom, E, Ep);
    edge_agg_kernel<<<cdiv((long long)Ep * (F >> 2), 256), 256, 0, stream>>>(
        src, dst, ex, denom, h, agg, E, Ep, F);
    relu_kernel<<<grd((long long)N * F), 256, 0, stream>>>(agg, (size_t)N * F);
  };

  auto run_side = [&](const float* x, const int* eidx, const int* batch,
                      const float* W1, const float* a1s, const float* a1d, const float* b1,
                      const float* W2, const float* a2s, const float* a2d, const float* b2,
                      float* pooled) {
    const int* src = eidx;
    const int* dst = eidx + E;
    gat_layer(x, src, dst, W1, a1s, a1d, b1, A, B, 256, 128);   // 256 -> 128
    gat_layer(B, src, dst, W2, a2s, a2d, b2, A, C, 128, 64);    // 128 -> 64
    fill_kernel<<<grd((long long)G * 64), 256, 0, stream>>>(sums, 0.f, (size_t)G * 64);
    fill_kernel<<<grd(G), 256, 0, stream>>>(cnt, 0.f, G);
    pool_sum_kernel<<<cdiv((long long)N * 64, 256), 256, 0, stream>>>(C, batch, sums, cnt, N, 64);
    pool_div_kernel<<<cdiv((long long)G * 64, 256), 256, 0, stream>>>(sums, cnt, pooled, G, 64);
  };

  run_side(x_s, ei_s, bs, Ws1, as1s, as1d, bs1, Ws2, as2s, as2d, bs2, pooled_s);
  run_side(x_t, ei_t, bt, Wt1, at1s, at1d, bt1, Wt2, at2s, at2d, bt2, pooled_t);

  classifier_kernel<<<cdiv((long long)G * L, 256), 256, 0, stream>>>(
      pooled_s, pooled_t, Wlin, blin, (float*)d_out, G, 64, L);
}
